// HybridQuantumDQN_70377334112277
// MI455X (gfx1250) — matrix operand order repaired and checked by reading, not compiled, not run on hardware
//
#include <hip/hip_runtime.h>
#include <hip/hip_bf16.h>

// ---------------------------------------------------------------------------
// Types for CDNA5 WMMA (wave32, 16x16x32 bf16 -> f32) and TDM descriptors
// ---------------------------------------------------------------------------
typedef __attribute__((ext_vector_type(16))) __bf16         v16bf;
typedef __attribute__((ext_vector_type(16))) unsigned short v16us;
typedef __attribute__((ext_vector_type(8)))  unsigned short v8us;
typedef __attribute__((ext_vector_type(8)))  float          v8f;
typedef __attribute__((ext_vector_type(4)))  unsigned int   u32x4;
typedef __attribute__((ext_vector_type(8)))  int            i32x8;
typedef __attribute__((ext_vector_type(4)))  int            i32x4;

union Frag16 {
    v16us v;
    struct { v8us lo; v8us hi; } p;
};

// ---------------------------------------------------------------------------
// bf16 <-> f32 helpers (round-to-nearest-even on pack)
// ---------------------------------------------------------------------------
__device__ __forceinline__ unsigned short f2bf(float f) {
    union { float f; unsigned int u; } v; v.f = f;
    unsigned int r = v.u + 0x7FFFu + ((v.u >> 16) & 1u);
    return (unsigned short)(r >> 16);
}
__device__ __forceinline__ float bf2f(unsigned short h) {
    union { unsigned int u; float f; } v; v.u = ((unsigned int)h) << 16;
    return v.f;
}

// ---------------------------------------------------------------------------
// Pack f32 weights [N, Kreal] -> bf16 [N, Kpad], zero-padded past Kreal.
// ---------------------------------------------------------------------------
__global__ void pack_w_kernel(const float* __restrict__ w,
                              unsigned short* __restrict__ out,
                              int N, int Kreal, int Kpad) {
    const int t = blockIdx.x * blockDim.x + threadIdx.x;
    if (t >= N * Kpad) return;
    const int n = t / Kpad, k = t - n * Kpad;
    out[t] = (k < Kreal) ? f2bf(w[(size_t)n * Kreal + k]) : (unsigned short)0;
}

// ---------------------------------------------------------------------------
// Implicit-GEMM conv:  out[b,n,oy,ox] = relu( sum_k A[r,k]*Wp[n,k] + bias[n] )
//   A[r,k] = in[b, c, oy*S+ky, ox*S+kx]   (gathered on the fly)
//   Wp = packed bf16 weights [N, K] (OIHW flattens to exactly n*K + k)
// One wave per 16x16 tile, 8 waves/block. ISA 7.12.2 lane maps:
//   A elem i -> K = k0 + (i/8)*16 + half*8 + (i%8)   (two contiguous runs of 8)
//   B elem i -> K = k0 + (i/8)*16 + half*8 + (i%8)   (two contiguous runs of 8)
//   D elem i -> M = m0 + half*8 + i, N = n0 + lane%16
// ---------------------------------------------------------------------------
template <bool IN_F32>
__global__ __launch_bounds__(256)
void conv_wmma_kernel(const void* __restrict__ in_,
                      const unsigned short* __restrict__ Wp,
                      const float* __restrict__ bias,
                      unsigned short* __restrict__ out,
                      int C, int H, int W, int KH, int KW, int S,
                      int OH, int OW, int N, int K, int M) {
    const int lane = threadIdx.x & 31;
    const int wid  = threadIdx.x >> 5;
    const int hf   = lane >> 4;
    const int ln   = lane & 15;
    const int m0   = (blockIdx.x * 8 + wid) * 16;
    const int n0   = blockIdx.y * 16;
    if (m0 >= M) return;

    const int ohw = OH * OW;
    const int khw = KH * KW;

    const int r   = m0 + ln;
    const int b   = r / ohw;
    const int rem = r - b * ohw;
    const int oy  = rem / OW;
    const int ox  = rem - oy * OW;

    const float*          inF = (const float*)in_;
    const unsigned short* inH = (const unsigned short*)in_;
    const int n = n0 + ln;
    const unsigned short* wrowBase = Wp + (size_t)n * K;

    v8f acc = {};
    for (int k0 = 0; k0 < K; k0 += 32) {
        Frag16 fa, fb;
        #pragma unroll
        for (int i = 0; i < 16; ++i) {
            const int k  = k0 + ((i >> 3) << 4) + hf * 8 + (i & 7);
            const int c  = k / khw;
            const int kr = k - c * khw;
            const int ky = kr / KW;
            const int kx = kr - ky * KW;
            const size_t idx = ((size_t)(b * C + c) * H + (oy * S + ky)) * W + (ox * S + kx);
            fa.v[i] = IN_F32 ? f2bf(inF[idx]) : inH[idx];
        }
        const unsigned short* wrow = wrowBase + k0 + hf * 8;
        fb.p.lo = *(const v8us*)(wrow);
        fb.p.hi = *(const v8us*)(wrow + 16);
        if (k0 + 32 < K) __builtin_prefetch(wrowBase + k0 + 32, 0, 1);

        v16bf av = __builtin_bit_cast(v16bf, fa.v);
        v16bf bv = __builtin_bit_cast(v16bf, fb.v);
        acc = __builtin_amdgcn_wmma_f32_16x16x32_bf16(
            false, av, false, bv, (short)0, acc, false, false);
    }

    const float bvv = bias[n];
    #pragma unroll
    for (int i = 0; i < 8; ++i) {
        const int m  = m0 + hf * 8 + i;
        const int rb = m / ohw;
        const int rr = m - rb * ohw;
        const int ry = rr / OW;
        const int rx = rr - ry * OW;
        float vv = acc[i] + bvv;
        vv = vv > 0.f ? vv : 0.f;
        out[((size_t)(rb * N + n) * OH + ry) * OW + rx] = f2bf(vv);
    }
}

// ---------------------------------------------------------------------------
// Dense GEMM with TDM-staged B panels:
//   out[m,n] = relu( sum_k A[m,k]*Wp[n,k] + bias[n] )
// All 8 waves of a block share n0, so the 16 x TK bf16 B-chunk is DMA'd into
// LDS once per block by wave 0 via TENSOR_LOAD_TO_LDS (D# per ISA ch.8),
// completion enforced with s_wait_tensorcnt + workgroup barrier, then B
// fragments are served from LDS (ds_load_b128).
// Grids are exact (M=1024, N,Kpad multiples of 16/32): no guards, no early
// returns (barrier-safe).
// ---------------------------------------------------------------------------
#define TKMAX 512
__global__ __launch_bounds__(256)
void fc_wmma_kernel(const unsigned short* __restrict__ A, int lda,
                    const unsigned short* __restrict__ Wp, // bf16 [N, Kpad]
                    const float* __restrict__ bias,
                    unsigned short* __restrict__ out, int ldo,
                    int M, int N, int Kpad) {
    __shared__ unsigned short sB[16 * TKMAX];   // 16 KB

    const int lane = threadIdx.x & 31;
    const int wid  = threadIdx.x >> 5;
    const int hf   = lane >> 4;
    const int ln   = lane & 15;
    const int m0   = (blockIdx.x * 8 + wid) * 16;
    const int n0   = blockIdx.y * 16;

    const unsigned short* arow = A + (size_t)(m0 + ln) * lda;

    v8f acc = {};
    for (int kc = 0; kc < Kpad; kc += TKMAX) {
        const int TK = (Kpad - kc < TKMAX) ? (Kpad - kc) : TKMAX;

        if (wid == 0) {
            // ---- Tensor DMA: 2D tile (16 rows of N) x (TK cols of K) ----
            const unsigned long long gaddr =
                (unsigned long long)(Wp + (size_t)n0 * Kpad + kc);
            const unsigned int ldsa = (unsigned int)(unsigned long long)&sB[0];

            u32x4 g0;
            g0[0] = 1u;                                   // count=1, user desc
            g0[1] = ldsa;                                 // lds_addr
            g0[2] = (unsigned int)(gaddr & 0xFFFFFFFFull);        // global_addr lo
            g0[3] = (unsigned int)((gaddr >> 32) & 0x01FFFFFFull) // global_addr hi
                    | (2u << 30);                         // type=2 ("image")

            i32x8 g1;
            g1[0] = (int)(1u << 16);                      // wg_mask=0, data_size=1 (2B)
            g1[1] = (int)(((unsigned)TK & 0xFFFFu) << 16);// tensor_dim0 lo16 @bit48
            g1[2] = (int)((((unsigned)TK >> 16) & 0xFFFFu)// tensor_dim0 hi16
                    | (16u << 16));                       // tensor_dim1 lo16 (=16)
            g1[3] = (int)(((unsigned)TK & 0xFFFFu) << 16);// dim1 hi16=0 | tile_dim0=TK
            g1[4] = 16;                                   // tile_dim1=16, tile_dim2=0
            g1[5] = (int)(unsigned)Kpad;                  // tensor_dim0_stride lo32
            g1[6] = 0;                                    // stride0 hi16 | stride1 lo16
            g1[7] = 0;                                    // stride1 hi32

            __builtin_amdgcn_tensor_load_to_lds(g0, g1, (i32x4)0, (i32x4)0,
                                                (i32x8)0, 0);
            __builtin_amdgcn_s_wait_tensorcnt(0);
        }
        __syncthreads();

        for (int k0 = kc; k0 < kc + TK; k0 += 32) {
            Frag16 fa, fb;
            fa.p.lo = *(const v8us*)(arow + k0 + hf * 8);
            fa.p.hi = *(const v8us*)(arow + k0 + 16 + hf * 8);
            const unsigned short* brow = sB + ln * TK + (k0 - kc) + hf * 8;
            fb.p.lo = *(const v8us*)(brow);
            fb.p.hi = *(const v8us*)(brow + 16);
            if (k0 + 32 < kc + TK) __builtin_prefetch(arow + k0 + 32, 0, 1);

            v16bf av = __builtin_bit_cast(v16bf, fa.v);
            v16bf bv = __builtin_bit_cast(v16bf, fb.v);
            acc = __builtin_amdgcn_wmma_f32_16x16x32_bf16(
                false, av, false, bv, (short)0, acc, false, false);
        }
        __syncthreads();
    }

    const int n = n0 + ln;
    const float bvv = bias[n];
    #pragma unroll
    for (int i = 0; i < 8; ++i) {
        const int m = m0 + hf * 8 + i;
        float vv = acc[i] + bvv;
        vv = vv > 0.f ? vv : 0.f;
        out[(size_t)m * ldo + n] = f2bf(vv);
    }
}

// ---------------------------------------------------------------------------
// qi = tanh(relu(feat@qi1^T)@qi2^T + b)  (tiny: 1024x6, K=64)
// ---------------------------------------------------------------------------
__global__ void qi2_tanh_kernel(const unsigned short* __restrict__ q1,
                                const float* __restrict__ w,
                                const float* __restrict__ b,
                                float* __restrict__ qi) {
    const int t = blockIdx.x * blockDim.x + threadIdx.x;
    if (t >= 1024 * 6) return;
    const int bi = t / 6, j = t - bi * 6;
    float s = b[j];
    for (int k = 0; k < 64; ++k) s += bf2f(q1[bi * 64 + k]) * w[j * 64 + k];
    qi[t] = tanhf(s);
}

// ---------------------------------------------------------------------------
// 6-qubit, 2-layer variational circuit; one thread per batch row.
// Qubit q lives on tensor axis q -> bit (5-q) of the flat amplitude index.
// ---------------------------------------------------------------------------
__global__ void quantum_kernel(const float* __restrict__ qi,
                               const float* __restrict__ params,
                               float* __restrict__ qf) {
    const int bi = blockIdx.x * blockDim.x + threadIdx.x;
    if (bi >= 1024) return;

    float sr[64], si[64];
    for (int s = 0; s < 64; ++s) { sr[s] = 0.f; si[s] = 0.f; }
    sr[0] = 1.f;
    const float PI = 3.14159265358979323846f;

    for (int q = 0; q < 6; ++q) {                 // encoding RY(inp*pi)
        const float th = 0.5f * qi[bi * 6 + q] * PI;
        const float c = cosf(th), s = sinf(th);
        const int mask = 1 << (5 - q);
        for (int st = 0; st < 64; ++st) {
            if (st & mask) continue;
            const int p = st | mask;
            const float a0r = sr[st], a0i = si[st], a1r = sr[p], a1i = si[p];
            sr[st] = c * a0r - s * a1r;  si[st] = c * a0i - s * a1i;
            sr[p]  = s * a0r + c * a1r;  si[p]  = s * a0i + c * a1i;
        }
    }
    for (int layer = 0; layer < 2; ++layer) {
        for (int q = 0; q < 5; ++q) {             // CNOT chain
            const int cm = 1 << (5 - q), tm = 1 << (4 - q);
            for (int st = 0; st < 64; ++st) {
                if ((st & cm) && !(st & tm)) {
                    const int p = st | tm;
                    const float tr = sr[st], ti = si[st];
                    sr[st] = sr[p]; si[st] = si[p];
                    sr[p]  = tr;    si[p]  = ti;
                }
            }
        }
        for (int q = 0; q < 6; ++q) {             // RY then RZ
            const int mask = 1 << (5 - q);
            const float thy = 0.5f * params[(layer * 6 + q) * 3 + 0];
            const float cy = cosf(thy), sy = sinf(thy);
            for (int st = 0; st < 64; ++st) {
                if (st & mask) continue;
                const int p = st | mask;
                const float a0r = sr[st], a0i = si[st], a1r = sr[p], a1i = si[p];
                sr[st] = cy * a0r - sy * a1r;  si[st] = cy * a0i - sy * a1i;
                sr[p]  = sy * a0r + cy * a1r;  si[p]  = sy * a0i + cy * a1i;
            }
            const float thz = 0.5f * params[(layer * 6 + q) * 3 + 1];
            const float cz = cosf(thz), sz = sinf(thz);
            for (int st = 0; st < 64; ++st) {
                const float r = sr[st], m = si[st];
                if (st & mask) { sr[st] = r * cz - m * sz; si[st] = m * cz + r * sz; }
                else           { sr[st] = r * cz + m * sz; si[st] = m * cz - r * sz; }
            }
        }
    }
    for (int q = 0; q < 6; ++q) {                 // <Z_q>
        const int mask = 1 << (5 - q);
        float e = 0.f;
        for (int st = 0; st < 64; ++st) {
            const float p = sr[st] * sr[st] + si[st] * si[st];
            e += (st & mask) ? -p : p;
        }
        qf[bi * 6 + q] = e;
    }
}

// ---------------------------------------------------------------------------
// comb[b, 0:512]=feat, [512:518]=qfeat, [518:544]=0   (K padded to 544)
// ---------------------------------------------------------------------------
__global__ void build_comb_kernel(const unsigned short* __restrict__ feat,
                                  const float* __restrict__ qf,
                                  unsigned short* __restrict__ comb) {
    const int t = blockIdx.x * blockDim.x + threadIdx.x;
    if (t >= 1024 * 544) return;
    const int bi = t / 544, c = t - bi * 544;
    unsigned short v;
    if (c < 512)      v = feat[bi * 512 + c];
    else if (c < 518) v = f2bf(qf[bi * 6 + (c - 512)]);
    else              v = 0;
    comb[t] = v;
}

// ---------------------------------------------------------------------------
// final head: out[b,j] = sum_k o1[b,k]*out2_w[j,k] + out2_b[j]
// ---------------------------------------------------------------------------
__global__ void out2_kernel(const unsigned short* __restrict__ o1,
                            const float* __restrict__ w,
                            const float* __restrict__ b,
                            float* __restrict__ out) {
    const int t = blockIdx.x * blockDim.x + threadIdx.x;
    if (t >= 1024 * 6) return;
    const int bi = t / 6, j = t - bi * 6;
    float s = b[j];
    for (int k = 0; k < 256; ++k) s += bf2f(o1[bi * 256 + k]) * w[j * 256 + k];
    out[t] = s;
}

// ---------------------------------------------------------------------------
// host launcher
// ---------------------------------------------------------------------------
extern "C" void kernel_launch(void* const* d_in, const int* in_sizes, int n_in,
                              void* d_out, int out_size, void* d_ws, size_t ws_size,
                              hipStream_t stream) {
    (void)in_sizes; (void)n_in; (void)out_size; (void)ws_size;
    const float* x       = (const float*)d_in[0];
    const float* conv1_w = (const float*)d_in[1];
    const float* conv1_b = (const float*)d_in[2];
    const float* conv2_w = (const float*)d_in[3];
    const float* conv2_b = (const float*)d_in[4];
    const float* conv3_w = (const float*)d_in[5];
    const float* conv3_b = (const float*)d_in[6];
    const float* fc_w    = (const float*)d_in[7];
    const float* fc_b    = (const float*)d_in[8];
    const float* qi1_w   = (const float*)d_in[9];
    const float* qi1_b   = (const float*)d_in[10];
    const float* qi2_w   = (const float*)d_in[11];
    const float* qi2_b   = (const float*)d_in[12];
    const float* qparams = (const float*)d_in[13];
    const float* out1_w  = (const float*)d_in[14];
    const float* out1_b  = (const float*)d_in[15];
    const float* out2_w  = (const float*)d_in[16];
    const float* out2_b  = (const float*)d_in[17];
    float* out = (float*)d_out;

    char* ws = (char*)d_ws;
    size_t off = 0;
    auto take = [&](size_t bytes) -> char* {
        char* p = ws + off;
        off = (off + bytes + 255) & ~(size_t)255;
        return p;
    };
    // activations
    unsigned short* h1   = (unsigned short*)take((size_t)1024 * 32 * 20 * 20 * 2);
    unsigned short* h2   = (unsigned short*)take((size_t)1024 * 64 * 9 * 9 * 2);
    unsigned short* h3   = (unsigned short*)take((size_t)1024 * 64 * 7 * 7 * 2);
    unsigned short* feat = (unsigned short*)take((size_t)1024 * 512 * 2);
    unsigned short* q1   = (unsigned short*)take((size_t)1024 * 64 * 2);
    float*          qi   = (float*)take((size_t)1024 * 6 * 4);
    float*          qf   = (float*)take((size_t)1024 * 6 * 4);
    unsigned short* comb = (unsigned short*)take((size_t)1024 * 544 * 2);
    unsigned short* o1   = (unsigned short*)take((size_t)1024 * 256 * 2);
    // packed bf16 weights
    unsigned short* w1p  = (unsigned short*)take((size_t)32 * 256 * 2);
    unsigned short* w2p  = (unsigned short*)take((size_t)64 * 512 * 2);
    unsigned short* w3p  = (unsigned short*)take((size_t)64 * 576 * 2);
    unsigned short* wfp  = (unsigned short*)take((size_t)512 * 3136 * 2);
    unsigned short* wq1p = (unsigned short*)take((size_t)64 * 512 * 2);
    unsigned short* wo1p = (unsigned short*)take((size_t)256 * 544 * 2);

    // ---- weight packing (f32 -> bf16, zero-padded K) ----
    pack_w_kernel<<<(32 * 256 + 255) / 256, 256, 0, stream>>>(conv1_w, w1p, 32, 256, 256);
    pack_w_kernel<<<(64 * 512 + 255) / 256, 256, 0, stream>>>(conv2_w, w2p, 64, 512, 512);
    pack_w_kernel<<<(64 * 576 + 255) / 256, 256, 0, stream>>>(conv3_w, w3p, 64, 576, 576);
    pack_w_kernel<<<(512 * 3136 + 255) / 256, 256, 0, stream>>>(fc_w, wfp, 512, 3136, 3136);
    pack_w_kernel<<<(64 * 512 + 255) / 256, 256, 0, stream>>>(qi1_w, wq1p, 64, 512, 512);
    pack_w_kernel<<<(256 * 544 + 255) / 256, 256, 0, stream>>>(out1_w, wo1p, 256, 518, 544);

    // ---- conv tower (implicit GEMM, WMMA) ----
    conv_wmma_kernel<true><<<dim3(409600 / 128, 2), 256, 0, stream>>>(
        (const void*)x, w1p, conv1_b, h1, 4, 84, 84, 8, 8, 4, 20, 20, 32, 256, 409600);
    conv_wmma_kernel<false><<<dim3(82944 / 128, 4), 256, 0, stream>>>(
        (const void*)h1, w2p, conv2_b, h2, 32, 20, 20, 4, 4, 2, 9, 9, 64, 512, 82944);
    conv_wmma_kernel<false><<<dim3(50176 / 128, 4), 256, 0, stream>>>(
        (const void*)h2, w3p, conv3_b, h3, 64, 9, 9, 3, 3, 1, 7, 7, 64, 576, 50176);

    // ---- dense layers (WMMA + TDM-staged B panels) ----
    fc_wmma_kernel<<<dim3(8, 32), 256, 0, stream>>>(h3, 3136, wfp, fc_b, feat, 512,
                                                    1024, 512, 3136);
    fc_wmma_kernel<<<dim3(8, 4), 256, 0, stream>>>(feat, 512, wq1p, qi1_b, q1, 64,
                                                   1024, 64, 512);
    qi2_tanh_kernel<<<(1024 * 6 + 255) / 256, 256, 0, stream>>>(q1, qi2_w, qi2_b, qi);
    quantum_kernel<<<1024 / 64, 64, 0, stream>>>(qi, qparams, qf);
    build_comb_kernel<<<(1024 * 544 + 255) / 256, 256, 0, stream>>>(feat, qf, comb);
    fc_wmma_kernel<<<dim3(8, 16), 256, 0, stream>>>(comb, 544, wo1p, out1_b, o1, 256,
                                                    1024, 256, 544);
    out2_kernel<<<(1024 * 6 + 255) / 256, 256, 0, stream>>>(o1, out2_w, out2_b, out);
}
